// GNN_76690936038144
// MI455X (gfx1250) — matrix operand order repaired and checked by reading, not compiled, not run on hardware
//
#include <hip/hip_runtime.h>

// ---------------- problem constants (fixed by the reference) ----------------
#define NNODES 4096      // B*N = 4*1024
#define CF     256       // feature dim (C == HID == 256)
#define OUTW   1024      // output row width: concat of 4 x 256
// GEMM per layer: Y[4096, 512] = Xb[4096,256](bf16) @ Wp[512,256]^T(bf16)
//   Y[:,0:256]   = x @ Wl^T              ("source" term)
//   Y[:,256:512] = x @ (Wr-Wl)^T         ("dest" term; bias added in edge pass)

typedef __bf16 bf16_t;
typedef __attribute__((ext_vector_type(16))) __bf16 v16bf;
typedef __attribute__((ext_vector_type(8)))  __bf16 v8bf;
typedef __attribute__((ext_vector_type(8)))  float  v8f;

__device__ __forceinline__ unsigned short f2bf(float f) {
  unsigned u = __float_as_uint(f);
  unsigned r = 0x7FFFu + ((u >> 16) & 1u);   // round-to-nearest-even
  return (unsigned short)((u + r) >> 16);
}

// ---- init: copy x0 into out[:,0:256], zero out[:,256:1024], x0 -> bf16 -----
__global__ void __launch_bounds__(256)
init_out_kernel(const float* __restrict__ x0, float* __restrict__ out,
                unsigned short* __restrict__ Xb) {
  int idx = blockIdx.x * blockDim.x + threadIdx.x;   // over 4096*1024
  int n = idx >> 10;
  int c = idx & (OUTW - 1);
  if (c < CF) {
    float v = x0[n * CF + c];
    out[idx] = v;
    Xb[n * CF + c] = f2bf(v);
  } else {
    out[idx] = 0.0f;
  }
}

// ---- weight prep: W[256,512] fp32 -> Wp[512,256] bf16 (Wl ; Wr-Wl) ---------
__global__ void __launch_bounds__(256)
prep_w_kernel(const float* __restrict__ W, unsigned short* __restrict__ Wp) {
  int idx = blockIdx.x * blockDim.x + threadIdx.x;   // over 256*256
  int o = idx >> 8;
  int k = idx & 255;
  float wl = W[o * 512 + k];
  float wr = W[o * 512 + 256 + k];
  Wp[o * 256 + k]         = f2bf(wl);
  Wp[(256 + o) * 256 + k] = f2bf(wr - wl);
}

// ---- WMMA GEMM: one wave computes a 16x64 tile of Y (A reused 4x) ----------
__global__ void __launch_bounds__(128)
gemm_bf16_kernel(const unsigned short* __restrict__ Xb,   // [4096,256] bf16
                 const unsigned short* __restrict__ Wpu,  // [512,256]  bf16
                 float* __restrict__ Y) {                 // [4096,512] f32
  const int lane = threadIdx.x & 31;
  const int wid  = (blockIdx.x * blockDim.x + threadIdx.x) >> 5;  // 0..2047
  const int mt   = wid >> 3;            // 0..255 : 16-row tile
  const int n0   = (wid & 7) << 6;      // 0..448 : 64-col group
  const int half = lane >> 4;           // 0/1
  const int rowl = lane & 15;

  const bf16_t* A  = reinterpret_cast<const bf16_t*>(Xb) + (mt * 16 + rowl) * 256;
  const bf16_t* Bp = reinterpret_cast<const bf16_t*>(Wpu);

  v8f acc[4] = {};

#pragma unroll
  for (int kk = 0; kk < 8; ++kk) {          // K = 256 in steps of 32
    const int k0 = kk * 32;
    // A 16x32 bf16 fragment: lane half selects 16 contiguous K values (matches B)
    union { v16bf v; v8bf h[2]; } a;
    a.v = *reinterpret_cast<const v16bf*>(A + k0 + half * 16);
#pragma unroll
    for (int t = 0; t < 4; ++t) {
      // B 32x16: lane holds column (n), 16 contiguous K values (32B load)
      const int col = n0 + t * 16 + rowl;
      v16bf b = *reinterpret_cast<const v16bf*>(Bp + col * 256 + k0 + half * 16);
      acc[t] = __builtin_amdgcn_wmma_f32_16x16x32_bf16(
          false, a.v, false, b, (short)0, acc[t], false, false);
    }
  }

  // C/D layout: VGPR v, lanes 0-15 -> M=v, lanes 16-31 -> M=8+v, N=lane%16
  const int mrow = mt * 16 + half * 8;
#pragma unroll
  for (int t = 0; t < 4; ++t) {
    const int col = n0 + t * 16 + rowl;
#pragma unroll
    for (int v = 0; v < 8; ++v)
      Y[(mrow + v) * 512 + col] = acc[t][v];
  }
}

// ---- edge pass: msg = relu(Yl[src]+Yr[dst]+b); atomic max into out slice ---
__global__ void __launch_bounds__(256)
edge_kernel(const float* __restrict__ Y,     // [4096,512]
            const float* __restrict__ bias,  // [256]
            const int* __restrict__ src, const int* __restrict__ dst,
            float* __restrict__ out,         // [4096,1024]
            int col_off, int nedges) {
  const int lane = threadIdx.x & 31;
  const int e = (blockIdx.x * blockDim.x + threadIdx.x) >> 5;  // one wave/edge
  if (e >= nedges) return;
  const int s = src[e];
  const int d = dst[e];
  const float4* yl = reinterpret_cast<const float4*>(Y + (size_t)s * 512) + lane * 2;
  const float4* yr = reinterpret_cast<const float4*>(Y + (size_t)d * 512 + 256) + lane * 2;
  const float4* bb = reinterpret_cast<const float4*>(bias) + lane * 2;
  unsigned int* op =
      reinterpret_cast<unsigned int*>(out + (size_t)d * OUTW + col_off) + lane * 8;
#pragma unroll
  for (int q = 0; q < 2; ++q) {
    float4 a = yl[q], r = yr[q], b = bb[q];
    // relu output >= 0 and slice zero-initialized: uint bit order == float order
    atomicMax(op + q * 4 + 0, __float_as_uint(fmaxf(a.x + r.x + b.x, 0.0f)));
    atomicMax(op + q * 4 + 1, __float_as_uint(fmaxf(a.y + r.y + b.y, 0.0f)));
    atomicMax(op + q * 4 + 2, __float_as_uint(fmaxf(a.z + r.z + b.z, 0.0f)));
    atomicMax(op + q * 4 + 3, __float_as_uint(fmaxf(a.w + r.w + b.w, 0.0f)));
  }
}

// ---- convert an output slice to bf16 as next layer's GEMM input ------------
__global__ void __launch_bounds__(256)
cvt_kernel(const float* __restrict__ out, int col_off,
           unsigned short* __restrict__ Xb) {
  int idx = blockIdx.x * blockDim.x + threadIdx.x;   // over 4096*256
  int n = idx >> 8;
  int c = idx & 255;
  Xb[idx] = f2bf(out[n * OUTW + col_off + c]);
}

extern "C" void kernel_launch(void* const* d_in, const int* in_sizes, int n_in,
                              void* d_out, int out_size, void* d_ws, size_t ws_size,
                              hipStream_t stream) {
  const float* x0 = (const float*)d_in[0];
  const int*   ei = (const int*)d_in[1];
  const float* W[3]  = {(const float*)d_in[2], (const float*)d_in[4], (const float*)d_in[6]};
  const float* bv[3] = {(const float*)d_in[3], (const float*)d_in[5], (const float*)d_in[7]};
  const int nedges = in_sizes[1] / 2;          // edge_index is [2, E]
  const int* src = ei;
  const int* dst = ei + nedges;
  float* out = (float*)d_out;

  // workspace layout: Xb (2MB) | Wp (256KB) | Y (8MB)
  char* ws = (char*)d_ws;
  unsigned short* Xb = (unsigned short*)ws;
  unsigned short* Wp = (unsigned short*)(ws + (size_t)NNODES * CF * 2);
  float*          Y  = (float*)(ws + (size_t)NNODES * CF * 2 + (size_t)512 * 256 * 2);

  init_out_kernel<<<(NNODES * OUTW) / 256, 256, 0, stream>>>(x0, out, Xb);

  for (int layer = 0; layer < 3; ++layer) {
    prep_w_kernel<<<(256 * 256) / 256, 256, 0, stream>>>(W[layer], Wp);
    gemm_bf16_kernel<<<512, 128, 0, stream>>>(Xb, Wp, Y);   // 2048 waves, 16x64 tiles
    edge_kernel<<<(nedges * 32 + 255) / 256, 256, 0, stream>>>(
        Y, bv[layer], src, dst, out, (layer + 1) * CF, nedges);
    if (layer < 2)
      cvt_kernel<<<(NNODES * CF) / 256, 256, 0, stream>>>(out, (layer + 1) * CF, Xb);
  }
}
